// GATModel_8203387535854
// MI455X (gfx1250) — matrix operand order repaired and checked by reading, not compiled, not run on hardware
//
#include <hip/hip_runtime.h>
#include <hip/hip_bf16.h>

typedef __bf16 bf16_t;
typedef __attribute__((ext_vector_type(8)))  bf16_t bf16x8;
typedef __attribute__((ext_vector_type(16))) bf16_t v16bf;
typedef __attribute__((ext_vector_type(8)))  float  v8f;

// ---------------------------------------------------------------------------
// f32 -> bf16 convert, optional per-column bias add + ReLU (fused inter-layer)
// ---------------------------------------------------------------------------
__global__ __launch_bounds__(256) void f32_to_bf16_k(
    const float* __restrict__ in, const float* __restrict__ bias,
    bf16_t* __restrict__ out, int total, int M, int do_relu)
{
    int i = blockIdx.x * blockDim.x + threadIdx.x;
    if (i >= total) return;
    float v = in[i];
    if (bias) v += bias[i % M];
    if (do_relu) v = fmaxf(v, 0.0f);
    out[i] = (bf16_t)v;
}

// W (K x M, row-major f32) -> Wt (M x K, row-major bf16)
__global__ __launch_bounds__(256) void wt_convert_k(
    const float* __restrict__ W, bf16_t* __restrict__ Wt, int K, int M)
{
    int i = blockIdx.x * blockDim.x + threadIdx.x;
    if (i >= K * M) return;
    int k = i / M, m = i - k * M;
    Wt[(size_t)m * K + k] = (bf16_t)W[i];
}

// ---------------------------------------------------------------------------
// GEMM: C[N x M] = A[N x K](bf16) * B[K x M] via Bt[M x K](bf16), f32 accum.
// One wave computes one 16x16 tile with v_wmma_f32_16x16x32_bf16.
// ---------------------------------------------------------------------------
__global__ __launch_bounds__(256) void gemm_bf16_wmma(
    const bf16_t* __restrict__ A, const bf16_t* __restrict__ Bt,
    float* __restrict__ C, int K, int M, int total_tiles, int tiles_col)
{
    const int lane = threadIdx.x & 31;
    const int tile = blockIdx.x * 8 + (threadIdx.x >> 5);
    if (tile >= total_tiles) return;                 // whole-wave uniform exit
    const int tile_row = tile / tiles_col;
    const int tile_col = tile - tile_row * tiles_col;

    const int hi   = (lane & 16) ? 1 : 0;           // upper half-wave
    const int asel = hi ? 8  : 0;                   // A: K 8-15 / 24-31 half
    const int bsel = hi ? 8  : 0;                   // B: K 8-15 / 24-31 half
    const bf16_t* arow = A  + (size_t)(tile_row * 16 + (lane & 15)) * K + asel;
    const bf16_t* brow = Bt + (size_t)(tile_col * 16 + (lane & 15)) * K + bsel;

    v8f acc = {};
    for (int k0 = 0; k0 < K; k0 += 32) {
        bf16x8 a0 = *(const bf16x8*)(arow + k0);        // K = k0+sel   .. +7
        bf16x8 a1 = *(const bf16x8*)(arow + k0 + 16);   // K = k0+sel+16.. +7
        bf16x8 b0 = *(const bf16x8*)(brow + k0);        // K = k0+bsel  .. +7
        bf16x8 b1 = *(const bf16x8*)(brow + k0 + 16);   // K = k0+bsel+16.. +7
        v16bf av = __builtin_shufflevector(a0, a1, 0,1,2,3,4,5,6,7,8,9,10,11,12,13,14,15);
        v16bf bv = __builtin_shufflevector(b0, b1, 0,1,2,3,4,5,6,7,8,9,10,11,12,13,14,15);
        acc = __builtin_amdgcn_wmma_f32_16x16x32_bf16(
                  false, av, false, bv, (short)0, acc, false, false);
    }

    // C/D layout: VGPR r -> row r (lanes 0-15) / row r+8 (lanes 16-31), col = lane&15
    float* cp = C + (size_t)(tile_row * 16 + (hi ? 8 : 0)) * M + tile_col * 16 + (lane & 15);
    #pragma unroll
    for (int r = 0; r < 8; ++r) cp[(size_t)r * M] = acc[r];
}

// ---------------------------------------------------------------------------
// Per-node attention logit halves: als[n,h] = <h[n,h,:], a_src[h,:]>, same dst
// ---------------------------------------------------------------------------
__global__ __launch_bounds__(256) void node_logits_k(
    const float* __restrict__ h, const float* __restrict__ a_src,
    const float* __restrict__ a_dst, float* __restrict__ als,
    float* __restrict__ ald, int N, int H, int C)
{
    int i = blockIdx.x * blockDim.x + threadIdx.x;
    if (i >= N * H) return;
    int n = i / H, hh = i - n * H;
    const float* hp = h + (size_t)n * H * C + (size_t)hh * C;
    const float* as = a_src + hh * C;
    const float* ad = a_dst + hh * C;
    float s = 0.f, d = 0.f;
    for (int c = 0; c < C; ++c) { float v = hp[c]; s += v * as[c]; d += v * ad[c]; }
    als[i] = s; ald[i] = d;
}

// order-preserving float<->uint mapping for atomic max
__device__ __forceinline__ unsigned f2ord(float f) {
    unsigned u = __float_as_uint(f);
    return (u & 0x80000000u) ? ~u : (u | 0x80000000u);
}
__device__ __forceinline__ float ord2f(unsigned u) {
    unsigned b = (u & 0x80000000u) ? (u & 0x7FFFFFFFu) : ~u;
    return __uint_as_float(b);
}

// Pass A: logit = leaky_relu(als[src]+ald[dst]); store; segment atomic max
__global__ __launch_bounds__(256) void edge_logit_max_k(
    const int* __restrict__ ei, int E, int Etot,
    const float* __restrict__ als, const float* __restrict__ ald,
    float* __restrict__ elog, unsigned* __restrict__ m, int H)
{
    int idx = blockIdx.x * blockDim.x + threadIdx.x;
    if (idx >= Etot * H) return;
    int e = idx / H, hh = idx - e * H;
    int src, dst;
    if (e < E) { src = ei[e]; dst = ei[E + e]; } else { src = dst = e - E; }
    float l = als[src * H + hh] + ald[dst * H + hh];
    l = (l > 0.f) ? l : 0.2f * l;
    elog[idx] = l;
    atomicMax(m + (size_t)dst * H + hh, f2ord(l));
}

// Pass B: e = exp(l - max[dst]); store over elog; segment atomic sum
__global__ __launch_bounds__(256) void edge_exp_sum_k(
    const int* __restrict__ ei, int E, int Etot,
    const unsigned* __restrict__ m, float* __restrict__ elog,
    float* __restrict__ denom, int H)
{
    int idx = blockIdx.x * blockDim.x + threadIdx.x;
    if (idx >= Etot * H) return;
    int e = idx / H, hh = idx - e * H;
    int dst = (e < E) ? ei[E + e] : (e - E);
    float ex = __expf(elog[idx] - ord2f(m[(size_t)dst * H + hh]));
    elog[idx] = ex;
    atomicAdd(denom + (size_t)dst * H + hh, ex);
}

// Pass C: one wave per edge; lane c covers channels c, c+32, ...
// agg[dst, ch] += h[src, ch] * (ex / denom[dst, head(ch)])
__global__ __launch_bounds__(256) void edge_scatter_k(
    const int* __restrict__ ei, int E, int Etot,
    const float* __restrict__ h, const float* __restrict__ elog,
    const float* __restrict__ denom, float* __restrict__ agg,
    int H, int HC, int cshift)
{
    int gt = blockIdx.x * blockDim.x + threadIdx.x;
    int e = gt >> 5, lane = gt & 31;
    if (e >= Etot) return;
    int src, dst;
    if (e < E) { src = ei[e]; dst = ei[E + e]; } else { src = dst = e - E; }
    for (int ch = lane; ch < HC; ch += 32) {
        int hh = ch >> cshift;
        float alpha = elog[(size_t)e * H + hh] / denom[(size_t)dst * H + hh];
        float v = h[(size_t)src * HC + ch] * alpha;
        atomicAdd(agg + (size_t)dst * HC + ch, v);
    }
}

// Final: out[c] = b3[c] + (1/N) * sum_n agg[n, c]   (M == 32)
__global__ void out_init_k(const float* __restrict__ b, float* __restrict__ out, int M)
{
    int i = threadIdx.x;
    if (i < M) out[i] = b[i];
}
__global__ __launch_bounds__(256) void mean_reduce_k(
    const float* __restrict__ agg, float* __restrict__ out, int N, float scale)
{
    int gt = blockIdx.x * blockDim.x + threadIdx.x;
    int c = gt & 31;
    int nstart = gt >> 5;
    int nstride = (int)((gridDim.x * blockDim.x) >> 5);
    float s = 0.f;
    for (int n = nstart; n < N; n += nstride) s += agg[(size_t)n * 32 + c];
    atomicAdd(out + c, s * scale);
}

// ---------------------------------------------------------------------------
static void gat_layer(const bf16_t* xin, const bf16_t* Wt,
                      const float* a_s, const float* a_d,
                      float* hbuf, float* als, float* ald,
                      unsigned* mb, float* dnm, float* elog, float* agg,
                      const int* ei, int N, int E, int Etot,
                      int K, int M, int H, int C, int cshift, hipStream_t stream)
{
    const int tiles_col = M / 16;
    const int total_tiles = (N / 16) * tiles_col;
    gemm_bf16_wmma<<<(total_tiles + 7) / 8, 256, 0, stream>>>(
        xin, Wt, hbuf, K, M, total_tiles, tiles_col);

    const int nh = N * H;
    node_logits_k<<<(nh + 255) / 256, 256, 0, stream>>>(hbuf, a_s, a_d, als, ald, N, H, C);

    hipMemsetAsync(mb,  0, (size_t)nh * 4, stream);
    hipMemsetAsync(dnm, 0, (size_t)nh * 4, stream);
    hipMemsetAsync(agg, 0, (size_t)N * M * 4, stream);

    const int eh = Etot * H;
    edge_logit_max_k<<<(eh + 255) / 256, 256, 0, stream>>>(ei, E, Etot, als, ald, elog, mb, H);
    edge_exp_sum_k  <<<(eh + 255) / 256, 256, 0, stream>>>(ei, E, Etot, mb, elog, dnm, H);
    const int st = Etot * 32;
    edge_scatter_k  <<<(st + 255) / 256, 256, 0, stream>>>(ei, E, Etot, hbuf, elog, dnm, agg,
                                                           H, M, cshift);
}

extern "C" void kernel_launch(void* const* d_in, const int* in_sizes, int n_in,
                              void* d_out, int out_size, void* d_ws, size_t ws_size,
                              hipStream_t stream)
{
    const float* x   = (const float*)d_in[0];
    const int*   ei  = (const int*)  d_in[1];
    const float* W1  = (const float*)d_in[2];
    const float* as1 = (const float*)d_in[3];
    const float* ad1 = (const float*)d_in[4];
    const float* b1  = (const float*)d_in[5];
    const float* W2  = (const float*)d_in[6];
    const float* as2 = (const float*)d_in[7];
    const float* ad2 = (const float*)d_in[8];
    const float* b2  = (const float*)d_in[9];
    const float* W3  = (const float*)d_in[10];
    const float* as3 = (const float*)d_in[11];
    const float* ad3 = (const float*)d_in[12];
    const float* b3  = (const float*)d_in[13];

    const int N    = in_sizes[0] / 128;
    const int E    = in_sizes[1] / 2;
    const int Etot = E + N;

    // workspace carve-out (256B aligned)
    char* ws = (char*)d_ws;
    size_t off = 0;
    auto take = [&](size_t bytes) -> char* {
        char* p = ws + off;
        off = (off + bytes + 255) & ~(size_t)255;
        return p;
    };
    bf16_t*   hbf  = (bf16_t*)  take((size_t)N * 128 * sizeof(bf16_t));
    float*    hbuf = (float*)   take((size_t)N * 128 * sizeof(float));
    float*    agg  = (float*)   take((size_t)N * 128 * sizeof(float));
    float*    als  = (float*)   take((size_t)N * 8 * sizeof(float));
    float*    ald  = (float*)   take((size_t)N * 8 * sizeof(float));
    unsigned* mb   = (unsigned*)take((size_t)N * 8 * sizeof(unsigned));
    float*    dnm  = (float*)   take((size_t)N * 8 * sizeof(float));
    float*    elog = (float*)   take((size_t)Etot * 8 * sizeof(float));
    bf16_t*   wt1  = (bf16_t*)  take((size_t)128 * 128 * sizeof(bf16_t));
    bf16_t*   wt2  = (bf16_t*)  take((size_t)128 * 128 * sizeof(bf16_t));
    bf16_t*   wt3  = (bf16_t*)  take((size_t)128 * 32  * sizeof(bf16_t));

    // transpose+convert weights to bf16 (tiny)
    wt_convert_k<<<(128 * 128 + 255) / 256, 256, 0, stream>>>(W1, wt1, 128, 128);
    wt_convert_k<<<(128 * 128 + 255) / 256, 256, 0, stream>>>(W2, wt2, 128, 128);
    wt_convert_k<<<(128 * 32  + 255) / 256, 256, 0, stream>>>(W3, wt3, 128, 32);

    // x -> bf16 GEMM operand
    const int nx = N * 128;
    f32_to_bf16_k<<<(nx + 255) / 256, 256, 0, stream>>>(x, nullptr, hbf, nx, 128, 0);

    // layer 1: 128 -> 8 heads x 16, concat
    gat_layer(hbf, wt1, as1, ad1, hbuf, als, ald, mb, dnm, elog, agg,
              ei, N, E, Etot, 128, 128, 8, 16, 4, stream);
    f32_to_bf16_k<<<(nx + 255) / 256, 256, 0, stream>>>(agg, b1, hbf, nx, 128, 1);

    // layer 2: 128 -> 8 heads x 16, concat
    gat_layer(hbf, wt2, as2, ad2, hbuf, als, ald, mb, dnm, elog, agg,
              ei, N, E, Etot, 128, 128, 8, 16, 4, stream);
    f32_to_bf16_k<<<(nx + 255) / 256, 256, 0, stream>>>(agg, b2, hbf, nx, 128, 1);

    // layer 3: 128 -> 1 head x 32 (concat=False, mean over 1 head == identity)
    gat_layer(hbf, wt3, as3, ad3, hbuf, als, ald, mb, dnm, elog, agg,
              ei, N, E, Etot, 128, 32, 1, 32, 5, stream);

    // global mean pool + bias
    out_init_k<<<1, 32, 0, stream>>>(b3, (float*)d_out, 32);
    mean_reduce_k<<<2048, 256, 0, stream>>>(agg, (float*)d_out, N, 1.0f / (float)N);
}
